// RangeLinearQuantParamLayerWrapper_13838384627736
// MI455X (gfx1250) — 8-bit matrix operand order repaired and checked by reading, compiled, not run on hardware
//
#include <hip/hip_runtime.h>
#include <cstdint>


typedef __attribute__((ext_vector_type(8))) int v8i;

#define M_DIM 8192
#define K_DIM 4096
#define N_DIM 4096

#define BLK_M 128
#define BLK_N 256
#define STAGE_A_BYTES (BLK_M * 64)
#define STAGE_B_BYTES (BLK_N * 64)
#define STAGE_BYTES   (STAGE_A_BYTES + STAGE_B_BYTES)

__global__ void init_scalars_kernel(unsigned* xmax_b, unsigned* wmax_b, int* amax_i) {
    *xmax_b = 0u;
    *wmax_b = 0u;
    *amax_i = 0;
}

__global__ __launch_bounds__(256) void absmax_kernel(const float4* __restrict__ src,
                                                     size_t n4, unsigned* out_bits) {
    float m = 0.0f;
    size_t idx = (size_t)blockIdx.x * blockDim.x + threadIdx.x;
    size_t stride = (size_t)gridDim.x * blockDim.x;
    for (size_t i = idx; i < n4; i += stride) {
        float4 v = src[i];
        m = fmaxf(m, fabsf(v.x));
        m = fmaxf(m, fabsf(v.y));
        m = fmaxf(m, fabsf(v.z));
        m = fmaxf(m, fabsf(v.w));
    }
    __shared__ float red[256];
    red[threadIdx.x] = m;
    __syncthreads();
    for (int s = 128; s > 0; s >>= 1) {
        if (threadIdx.x < s) red[threadIdx.x] = fmaxf(red[threadIdx.x], red[threadIdx.x + s]);
        __syncthreads();
    }
    if (threadIdx.x == 0) atomicMax(out_bits, __float_as_uint(red[0]));
}

__global__ __launch_bounds__(256) void quant_kernel(const float4* __restrict__ src,
                                                    int* __restrict__ dst, size_t n4,
                                                    const unsigned* __restrict__ max_bits) {
    const float scale = 127.0f / __uint_as_float(*max_bits);
    size_t idx = (size_t)blockIdx.x * blockDim.x + threadIdx.x;
    size_t stride = (size_t)gridDim.x * blockDim.x;
    for (size_t i = idx; i < n4; i += stride) {
        float4 v = src[i];
        int q0 = (int)fminf(fmaxf(rintf(v.x * scale), -128.0f), 127.0f);
        int q1 = (int)fminf(fmaxf(rintf(v.y * scale), -128.0f), 127.0f);
        int q2 = (int)fminf(fmaxf(rintf(v.z * scale), -128.0f), 127.0f);
        int q3 = (int)fminf(fmaxf(rintf(v.w * scale), -128.0f), 127.0f);
        dst[i] = (q0 & 0xff) | ((q1 & 0xff) << 8) | ((q2 & 0xff) << 16) | (q3 << 24);
    }
}

__global__ __launch_bounds__(256) void gemm_iu8_wmma_kernel(
    const signed char* __restrict__ Aq,
    const signed char* __restrict__ Bq,
    int* __restrict__ Cout,
    int* __restrict__ amax_out) {
    __shared__ signed char lds[2 * STAGE_BYTES];
    __shared__ int red[256];

    const int t     = threadIdx.x;
    const int lane  = t & 31;
    const int wave  = t >> 5;
    const int waveM = (wave >> 2) * 64;
    const int waveN = (wave & 3) * 64;
    const int half  = lane >> 4;
    const int l16   = lane & 15;

    const int blockM = blockIdx.y * BLK_M;
    const int blockN = blockIdx.x * BLK_N;

    v8i acc[4][4];
#pragma unroll
    for (int i = 0; i < 4; ++i)
#pragma unroll
        for (int j = 0; j < 4; ++j)
#pragma unroll
            for (int r = 0; r < 8; ++r) acc[i][j][r] = 0;

    auto stage = [&](int buf, int k0) {
        const unsigned base = (unsigned)(uintptr_t)&lds[buf * STAGE_BYTES];
#pragma unroll
        for (int s = 0; s < 2; ++s) {
            int linear = s * 256 + t;
            int row = linear >> 2;
            int ko  = (linear & 3) * 16;
            unsigned dst = base + (unsigned)(row * 64 + ko);
            const signed char* src = Aq + (size_t)(blockM + row) * K_DIM + k0 + ko;
            asm volatile("global_load_async_to_lds_b128 %0, %1, off"
                         :: "v"(dst), "v"(src) : "memory");
        }
#pragma unroll
        for (int s = 0; s < 4; ++s) {
            int linear = s * 256 + t;
            int row = linear >> 2;
            int ko  = (linear & 3) * 16;
            unsigned dst = base + (unsigned)(STAGE_A_BYTES + row * 64 + ko);
            const signed char* src = Bq + (size_t)(blockN + row) * K_DIM + k0 + ko;
            asm volatile("global_load_async_to_lds_b128 %0, %1, off"
                         :: "v"(dst), "v"(src) : "memory");
        }
    };

    stage(0, 0);
    int cur = 0;

    for (int k0 = 0; k0 < K_DIM; k0 += 64) {
        if (k0 + 64 < K_DIM) {
            stage(cur ^ 1, k0 + 64);
            asm volatile("s_wait_asynccnt 6" ::: "memory");
        } else {
            asm volatile("s_wait_asynccnt 0" ::: "memory");
        }
        __syncthreads();

        const signed char* As = &lds[cur * STAGE_BYTES];
        const signed char* Bs = As + STAGE_A_BYTES;

        v8i a[4];
#pragma unroll
        for (int i = 0; i < 4; ++i) {
            const signed char* ap = As + (waveM + i * 16 + l16) * 64 + half * 16;
            int2 c0 = *(const int2*)(ap +  0);
            int2 c1 = *(const int2*)(ap +  8);
            int2 c2 = *(const int2*)(ap + 32);
            int2 c3 = *(const int2*)(ap + 40);
            a[i][0] = c0.x; a[i][1] = c0.y;
            a[i][2] = c1.x; a[i][3] = c1.y;
            a[i][4] = c2.x; a[i][5] = c2.y;
            a[i][6] = c3.x; a[i][7] = c3.y;
        }
        v8i b[4];
#pragma unroll
        for (int j = 0; j < 4; ++j) {
            const signed char* bp = Bs + (waveN + j * 16 + l16) * 64 + half * 16;
            int4 q0 = *(const int4*)(bp +  0);
            int4 q1 = *(const int4*)(bp + 32);
            b[j][0] = q0.x; b[j][1] = q0.y; b[j][2] = q0.z; b[j][3] = q0.w;
            b[j][4] = q1.x; b[j][5] = q1.y; b[j][6] = q1.z; b[j][7] = q1.w;
        }
#pragma unroll
        for (int i = 0; i < 4; ++i)
#pragma unroll
            for (int j = 0; j < 4; ++j)
                acc[i][j] = __builtin_amdgcn_wmma_i32_16x16x64_iu8(
                     true, a[i],  true, b[j], acc[i][j],
                     false,  false);

        __syncthreads();
        cur ^= 1;
    }

    int localMax = 0;
#pragma unroll
    for (int i = 0; i < 4; ++i) {
#pragma unroll
        for (int j = 0; j < 4; ++j) {
            const int col = blockN + waveN + j * 16 + l16;
#pragma unroll
            for (int r = 0; r < 8; ++r) {
                const int row = blockM + waveM + i * 16 + half * 8 + r;
                int v = acc[i][j][r];
                Cout[(size_t)row * N_DIM + col] = v;
                int av = v < 0 ? -v : v;
                localMax = av > localMax ? av : localMax;
            }
        }
    }
    red[t] = localMax;
    __syncthreads();
    for (int s = 128; s > 0; s >>= 1) {
        if (t < s) {
            int o = red[t + s];
            if (o > red[t]) red[t] = o;
        }
        __syncthreads();
    }
    if (t == 0) atomicMax(amax_out, red[0]);
}

__global__ __launch_bounds__(256) void requant_kernel(int4* __restrict__ accum,
                                                      float4* __restrict__ out,
                                                      const unsigned* __restrict__ xmax_b,
                                                      const unsigned* __restrict__ wmax_b,
                                                      const int* __restrict__ amax_i,
                                                      size_t n4) {
    const float in_scale = 127.0f / __uint_as_float(*xmax_b);
    const float w_scale  = 127.0f / __uint_as_float(*wmax_b);
    const float amax     = (float)(*amax_i);
    const float rq       = 127.0f / amax;
    const float dq       = amax / (127.0f * in_scale * w_scale);
    size_t idx = (size_t)blockIdx.x * blockDim.x + threadIdx.x;
    size_t stride = (size_t)gridDim.x * blockDim.x;
    for (size_t i = idx; i < n4; i += stride) {
        int4 av = accum[i];
        float4 o;
        o.x = fminf(fmaxf(rintf((float)av.x * rq), -128.0f), 127.0f) * dq;
        o.y = fminf(fmaxf(rintf((float)av.y * rq), -128.0f), 127.0f) * dq;
        o.z = fminf(fmaxf(rintf((float)av.z * rq), -128.0f), 127.0f) * dq;
        o.w = fminf(fmaxf(rintf((float)av.w * rq), -128.0f), 127.0f) * dq;
        out[i] = o;
    }
}

extern "C" void kernel_launch(void* const* d_in, const int* in_sizes, int n_in,
                              void* d_out, int out_size, void* d_ws, size_t ws_size,
                              hipStream_t stream) {
    const float* x = (const float*)d_in[0];
    const float* W = (const float*)d_in[1];

    char* ws = (char*)d_ws;
    unsigned* xmax_b = (unsigned*)(ws + 0);
    unsigned* wmax_b = (unsigned*)(ws + 4);
    int*      amax_i = (int*)(ws + 8);
    signed char* xq = (signed char*)(ws + 64);
    signed char* wq = xq + (size_t)M_DIM * K_DIM;

    const size_t nx = (size_t)M_DIM * K_DIM;
    const size_t nw = (size_t)K_DIM * N_DIM;
    const size_t no = (size_t)M_DIM * N_DIM;

    init_scalars_kernel<<<1, 1, 0, stream>>>(xmax_b, wmax_b, amax_i);

    absmax_kernel<<<2048, 256, 0, stream>>>((const float4*)x, nx / 4, xmax_b);
    absmax_kernel<<<1024, 256, 0, stream>>>((const float4*)W, nw / 4, wmax_b);

    quant_kernel<<<4096, 256, 0, stream>>>((const float4*)x, (int*)xq, nx / 4, xmax_b);
    quant_kernel<<<2048, 256, 0, stream>>>((const float4*)W, (int*)wq, nw / 4, wmax_b);

    dim3 grid(N_DIM / BLK_N, M_DIM / BLK_M);
    gemm_iu8_wmma_kernel<<<grid, 256, 0, stream>>>(xq, wq, (int*)d_out, amax_i);

    requant_kernel<<<4096, 256, 0, stream>>>((int4*)d_out, (float4*)d_out,
                                             xmax_b, wmax_b, amax_i, no / 4);
}
